// Block_44873818308807
// MI455X (gfx1250) — compile-verified, matrix operand order checked by reading, not run on hardware
//
#include <hip/hip_runtime.h>
#include <hip/hip_bf16.h>
#include <math.h>

// Problem constants (from reference)
#define B_ 16
#define S_ 1024
#define E_ 768
#define H_ 12
#define D_ 64
#define F_ 3072
#define M_ (B_ * S_)   // 16384 tokens

typedef __attribute__((ext_vector_type(16))) _Float16 v16h;
typedef __attribute__((ext_vector_type(8)))  float    v8f;
typedef __attribute__((ext_vector_type(4)))  int      v4i_;

// Pointer types matching the async-LDS builtin signature (clang expects
// int4 pointers: global = AS(1), LDS = AS(3)).
typedef __attribute__((address_space(1))) v4i_* gptr_v4i;
typedef __attribute__((address_space(3))) v4i_* lptr_v4i;

// Async global->LDS path (CDNA5): guarded so a missing builtin degrades to
// the load+ds_store staging path instead of failing the build.
#if defined(__has_builtin)
#  if __has_builtin(__builtin_amdgcn_global_load_async_to_lds_b128)
#    define HAVE_ASYNC_LDS 1
#  else
#    define HAVE_ASYNC_LDS 0
#  endif
#  if __has_builtin(__builtin_amdgcn_s_wait_asynccnt)
#    define WAIT_ASYNCCNT0() __builtin_amdgcn_s_wait_asynccnt(0)
#  else
#    define WAIT_ASYNCCNT0() asm volatile("s_wait_asynccnt 0" ::: "memory")
#  endif
#else
#  define HAVE_ASYNC_LDS 0
#  define WAIT_ASYNCCNT0() asm volatile("s_wait_asynccnt 0" ::: "memory")
#endif

union FragH { v16h v; unsigned int u[8]; };

// Load a 16x32 f16 fragment (A-layout) from a row-major f16 matrix.
// Lane l holds row (row0 + l%16); K pairs at {0,2,4,6,16,18,20,22} + 8*(l/16).
__device__ __forceinline__ void load_frag(FragH& f, const _Float16* p,
                                          int row0, int ld, int k0) {
  const int lane = threadIdx.x & 31;
  const _Float16* base =
      p + (size_t)(row0 + (lane & 15)) * ld + k0 + ((lane >> 4) & 1) * 8;
#pragma unroll
  for (int j = 0; j < 8; ++j) {
    const int K = (j < 4) ? (2 * j) : (16 + 2 * (j - 4));
    f.u[j] = *(const unsigned int*)(base + K);
  }
}

__device__ __forceinline__ v8f wmma_f16(const FragH& a, const FragH& b, v8f c) {
  return __builtin_amdgcn_wmma_f32_16x16x32_f16(
      /*neg_a=*/false, a.v, /*neg_b=*/false, b.v,
      /*c_mod=*/(short)0, c, /*reuse_a=*/false, /*reuse_b=*/false);
}

// ---------------------------------------------------------------------------
// Weight transpose + f32->f16 convert: src [batch,K,N] f32 -> dst [batch,N,K] f16
// ---------------------------------------------------------------------------
__global__ void transpose_to_f16_kernel(const float* __restrict__ src,
                                        _Float16* __restrict__ dst,
                                        int batch, int K, int N) {
  size_t i = (size_t)blockIdx.x * blockDim.x + threadIdx.x;
  size_t total = (size_t)batch * K * N;
  if (i >= total) return;
  int k = (int)(i % K);
  size_t t = i / K;
  int n = (int)(t % N);
  int bt = (int)(t / N);
  dst[i] = (_Float16)src[((size_t)bt * K + k) * N + n];
}

// ---------------------------------------------------------------------------
// LayerNorm over E=768. mode 0: outH = f16(ln(x))   mode 1: outF = resid + ln(x)
// ---------------------------------------------------------------------------
__global__ __launch_bounds__(256)
void ln_kernel(const float* __restrict__ x, const float* __restrict__ g,
               const float* __restrict__ bta, const float* __restrict__ resid,
               _Float16* __restrict__ outH, float* __restrict__ outF, int mode) {
  const int row = blockIdx.x;
  const int tid = threadIdx.x;
  const float* xr = x + (size_t)row * E_;
  __shared__ float sS[256], sQ[256];
  float a = 0.f, q = 0.f;
  for (int e = tid; e < E_; e += 256) {
    float v = xr[e];
    a += v; q += v * v;
  }
  sS[tid] = a; sQ[tid] = q;
  __syncthreads();
  for (int off = 128; off > 0; off >>= 1) {
    if (tid < off) { sS[tid] += sS[tid + off]; sQ[tid] += sQ[tid + off]; }
    __syncthreads();
  }
  const float mu = sS[0] * (1.0f / E_);
  const float var = sQ[0] * (1.0f / E_) - mu * mu;
  const float rstd = rsqrtf(var + 1e-5f);
  for (int e = tid; e < E_; e += 256) {
    float y = (xr[e] - mu) * rstd * g[e] + bta[e];
    size_t i = (size_t)row * E_ + e;
    if (mode == 0) outH[i] = (_Float16)y;
    else           outF[i] = resid[i] + y;
  }
}

// ---------------------------------------------------------------------------
// WMMA GEMM: C[M,N] = A[M,K]f16 * Bt[N,K]f16^T + bias, fused epilogues.
// Block tile 128x64, 8 waves, each wave -> 16 rows x 64 cols (4 accum frags).
// B tile staged to LDS via GLOBAL_LOAD_ASYNC_TO_LDS_B128 when available.
// epi: 1 = scatter f16 to [B,H,S,D]   2 = scatter f16 to [B,H,D,S]
//      3 = outF = resid + C + bias    4 = outH = f16(gelu(C + bias))
//      5 = outF = C + bias
// ---------------------------------------------------------------------------
__global__ __launch_bounds__(256)
void gemm_wmma_kernel(const _Float16* __restrict__ A,
                      const _Float16* __restrict__ Bt,
                      const float* __restrict__ bias,
                      const float* __restrict__ resid,
                      float* __restrict__ outF, _Float16* __restrict__ outH,
                      int M, int N, int K, int epi, int Hh, int Dd, int Ss) {
  const int m0 = blockIdx.x * 128;
  const int n0 = blockIdx.y * 64;
  const int wave = threadIdx.x >> 5;
  const int lane = threadIdx.x & 31;
  const int mw = m0 + wave * 16;

  __shared__ _Float16 Bt_s[64][32];

  v8f c[4] = {};

  for (int kk = 0; kk < K; kk += 32) {
    __syncthreads();
    {  // cooperative B-tile stage: 64 rows x 32 halves = 4KB, 16B per thread
      const int rown = threadIdx.x >> 2;
      const int ko = (threadIdx.x & 3) * 8;
      const _Float16* gsrc = &Bt[(size_t)(n0 + rown) * K + kk + ko];
      _Float16* ldst = &Bt_s[rown][ko];
#if HAVE_ASYNC_LDS
      // Direct global->LDS DMA, no VGPR round trip (tracked by ASYNCcnt).
      // Global flat address == AS(1) address; LDS offset = addr[31:0].
      __builtin_amdgcn_global_load_async_to_lds_b128(
          (gptr_v4i)(uintptr_t)gsrc,
          (lptr_v4i)(unsigned int)(uintptr_t)ldst,
          0, 0);
      WAIT_ASYNCCNT0();
#else
      *(uint4*)ldst = *(const uint4*)gsrc;
#endif
      if (kk + 32 < K) __builtin_prefetch(gsrc + 32, 0, 1);
    }
    __syncthreads();
    FragH af;
    load_frag(af, A, mw, K, kk);
#pragma unroll
    for (int c4 = 0; c4 < 4; ++c4) {
      FragH bf;
      load_frag(bf, &Bt_s[0][0], c4 * 16, 32, 0);
      c[c4] = wmma_f16(af, bf, c[c4]);
    }
  }

#pragma unroll
  for (int c4 = 0; c4 < 4; ++c4) {
#pragma unroll
    for (int r = 0; r < 8; ++r) {
      const int row = mw + r + 8 * ((lane >> 4) & 1);
      const int col = n0 + c4 * 16 + (lane & 15);
      float val = c[c4][r] + bias[col];
      if (epi == 1) {
        int bb = row / Ss, ss = row % Ss, hh = col / Dd, dd = col % Dd;
        outH[(((size_t)(bb * Hh + hh)) * Ss + ss) * Dd + dd] = (_Float16)val;
      } else if (epi == 2) {
        int bb = row / Ss, ss = row % Ss, hh = col / Dd, dd = col % Dd;
        outH[(((size_t)(bb * Hh + hh)) * Dd + dd) * Ss + ss] = (_Float16)val;
      } else if (epi == 3) {
        size_t i = (size_t)row * N + col;
        outF[i] = resid[i] + val;
      } else if (epi == 4) {
        float gl = 0.5f * val * (1.0f + erff(val * 0.70710678118f));
        outH[(size_t)row * N + col] = (_Float16)gl;
      } else {
        outF[(size_t)row * N + col] = val;
      }
    }
  }
}

// ---------------------------------------------------------------------------
// Flash attention: one wave per 16-row query tile, online softmax over
// 32-key causal windows. q/k: [B,H,S,D] f16, vT: [B,H,D,S] f16,
// output o: [B,S,E] f16 (heads concatenated).
// ---------------------------------------------------------------------------
__global__ __launch_bounds__(128)
void attn_kernel(const _Float16* __restrict__ q, const _Float16* __restrict__ k,
                 const _Float16* __restrict__ vT, _Float16* __restrict__ op) {
  const int bh = blockIdx.x;
  const int bb = bh / H_;
  const int hh = bh % H_;
  const int wave = threadIdx.x >> 5;
  const int lane = threadIdx.x & 31;
  const int q0 = (blockIdx.y * 4 + wave) * 16;

  const _Float16* qh = q + (size_t)bh * S_ * D_;
  const _Float16* kh = k + (size_t)bh * S_ * D_;
  const _Float16* vh = vT + (size_t)bh * D_ * S_;

  __shared__ _Float16 Pt[4][16][40];  // per-wave P transpose staging

  FragH qf0, qf1;
  load_frag(qf0, qh, q0, D_, 0);
  load_frag(qf1, qh, q0, D_, 32);

  float m_[8], l_[8];
  v8f oc[4] = {};
#pragma unroll
  for (int r = 0; r < 8; ++r) { m_[r] = -1e30f; l_[r] = 0.f; }

  const int colb = lane & 15;
  const int lh = (lane >> 4) & 1;
  const float rsD = 0.125f;  // 1/sqrt(64)
  const int nwin = (q0 + 47) / 32;  // causal: last key <= q0+15

  for (int j = 0; j < nwin; ++j) {
    const int t0 = j * 32;
    v8f s0 = {}, s1 = {};
    FragH kf;
    load_frag(kf, kh, t0, D_, 0);        s0 = wmma_f16(qf0, kf, s0);
    load_frag(kf, kh, t0, D_, 32);       s0 = wmma_f16(qf1, kf, s0);
    load_frag(kf, kh, t0 + 16, D_, 0);   s1 = wmma_f16(qf0, kf, s1);
    load_frag(kf, kh, t0 + 16, D_, 32);  s1 = wmma_f16(qf1, kf, s1);

    float tmax[8];
#pragma unroll
    for (int r = 0; r < 8; ++r) {
      const int srow = q0 + r + 8 * lh;
      float a0 = ((t0 + colb)      <= srow) ? s0[r] * rsD : -1e30f;
      float a1 = ((t0 + 16 + colb) <= srow) ? s1[r] * rsD : -1e30f;
      s0[r] = a0; s1[r] = a1;
      float mx = fmaxf(a0, a1);
#pragma unroll
      for (int ofs = 1; ofs < 16; ofs <<= 1)
        mx = fmaxf(mx, __shfl_xor(mx, ofs, 16));
      tmax[r] = mx;
    }
#pragma unroll
    for (int r = 0; r < 8; ++r) {
      const float newm = fmaxf(m_[r], tmax[r]);
      const float alpha = __expf(m_[r] - newm);
      const float p0 = __expf(s0[r] - newm);
      const float p1 = __expf(s1[r] - newm);
      float rs = p0 + p1;
#pragma unroll
      for (int ofs = 1; ofs < 16; ofs <<= 1) rs += __shfl_xor(rs, ofs, 16);
      l_[r] = l_[r] * alpha + rs;
      m_[r] = newm;
#pragma unroll
      for (int c4 = 0; c4 < 4; ++c4) oc[c4][r] *= alpha;
      Pt[wave][r + 8 * lh][colb]      = (_Float16)p0;
      Pt[wave][r + 8 * lh][colb + 16] = (_Float16)p1;
    }
    asm volatile("s_wait_dscnt 0" ::: "memory");
    FragH pf;
    load_frag(pf, &Pt[wave][0][0], 0, 40, 0);
#pragma unroll
    for (int c4 = 0; c4 < 4; ++c4) {
      FragH vf;
      load_frag(vf, vh, c4 * 16, S_, t0);
      oc[c4] = wmma_f16(pf, vf, oc[c4]);
    }
  }

#pragma unroll
  for (int r = 0; r < 8; ++r) {
    const float inv = 1.0f / l_[r];
    const int srow = q0 + r + 8 * lh;
    const size_t basei = ((size_t)(bb * S_ + srow)) * E_ + hh * D_;
#pragma unroll
    for (int c4 = 0; c4 < 4; ++c4)
      op[basei + c4 * 16 + colb] = (_Float16)(oc[c4][r] * inv);
  }
}

// ---------------------------------------------------------------------------
extern "C" void kernel_launch(void* const* d_in, const int* in_sizes, int n_in,
                              void* d_out, int out_size, void* d_ws, size_t ws_size,
                              hipStream_t stream) {
  const float* x     = (const float*)d_in[0];
  const float* Wq    = (const float*)d_in[1];
  const float* bq    = (const float*)d_in[2];
  const float* Wk    = (const float*)d_in[3];
  const float* bk    = (const float*)d_in[4];
  const float* Wv    = (const float*)d_in[5];
  const float* bv    = (const float*)d_in[6];
  const float* Wp    = (const float*)d_in[7];
  const float* bp    = (const float*)d_in[8];
  const float* ln1_g = (const float*)d_in[9];
  const float* ln1_b = (const float*)d_in[10];
  const float* ln2_g = (const float*)d_in[11];
  const float* ln2_b = (const float*)d_in[12];
  const float* W1    = (const float*)d_in[13];
  const float* b1    = (const float*)d_in[14];
  const float* W2    = (const float*)d_in[15];
  const float* b2    = (const float*)d_in[16];
  const float* lnf_g = (const float*)d_in[17];
  const float* lnf_b = (const float*)d_in[18];
  float* out = (float*)d_out;

  char* ws = (char*)d_ws;
  size_t off = 0;
  auto alloc = [&](size_t bytes) -> void* {
    void* p = ws + off;
    off = (off + bytes + 255) & ~(size_t)255;
    return p;
  };
  _Float16* h16  = (_Float16*)alloc((size_t)M_ * E_ * 2);
  _Float16* q16  = (_Float16*)alloc((size_t)M_ * E_ * 2);
  _Float16* k16  = (_Float16*)alloc((size_t)M_ * E_ * 2);
  _Float16* vT16 = (_Float16*)alloc((size_t)M_ * E_ * 2);
  _Float16* o16  = (_Float16*)alloc((size_t)M_ * E_ * 2);
  float*    x1   = (float*)   alloc((size_t)M_ * E_ * 4);
  _Float16* h2   = (_Float16*)alloc((size_t)M_ * E_ * 2);
  _Float16* g16  = (_Float16*)alloc((size_t)M_ * F_ * 2);
  float*    fbuf = (float*)   alloc((size_t)M_ * E_ * 4);
  _Float16* WqT  = (_Float16*)alloc((size_t)H_ * D_ * E_ * 2);
  _Float16* WkT  = (_Float16*)alloc((size_t)H_ * D_ * E_ * 2);
  _Float16* WvT  = (_Float16*)alloc((size_t)H_ * D_ * E_ * 2);
  _Float16* WpT  = (_Float16*)alloc((size_t)E_ * E_ * 2);
  _Float16* W1T  = (_Float16*)alloc((size_t)F_ * E_ * 2);
  _Float16* W2T  = (_Float16*)alloc((size_t)E_ * F_ * 2);

  auto tlaunch = [&](const float* src, _Float16* dst, int batch, int K, int N) {
    size_t total = (size_t)batch * K * N;
    transpose_to_f16_kernel<<<dim3((unsigned)((total + 255) / 256)), 256, 0, stream>>>(
        src, dst, batch, K, N);
  };
  // weights -> [N,K] f16
  tlaunch(Wq, WqT, H_, E_, D_);
  tlaunch(Wk, WkT, H_, E_, D_);
  tlaunch(Wv, WvT, H_, E_, D_);
  tlaunch(Wp, WpT, 1, E_, E_);
  tlaunch(W1, W1T, 1, E_, F_);
  tlaunch(W2, W2T, 1, F_, E_);

  // h = ln1(x) -> f16
  ln_kernel<<<M_, 256, 0, stream>>>(x, ln1_g, ln1_b, nullptr, h16, nullptr, 0);

  // q/k/v projections (N = H*D = 768, K = 768)
  gemm_wmma_kernel<<<dim3(M_ / 128, E_ / 64), 256, 0, stream>>>(
      h16, WqT, bq, nullptr, nullptr, q16, M_, E_, E_, 1, H_, D_, S_);
  gemm_wmma_kernel<<<dim3(M_ / 128, E_ / 64), 256, 0, stream>>>(
      h16, WkT, bk, nullptr, nullptr, k16, M_, E_, E_, 1, H_, D_, S_);
  gemm_wmma_kernel<<<dim3(M_ / 128, E_ / 64), 256, 0, stream>>>(
      h16, WvT, bv, nullptr, nullptr, vT16, M_, E_, E_, 2, H_, D_, S_);

  // causal flash attention -> o16 [B,S,E]
  attn_kernel<<<dim3(B_ * H_, S_ / 64), 128, 0, stream>>>(q16, k16, vT16, o16);

  // x1 = x + o @ Wp + bp
  gemm_wmma_kernel<<<dim3(M_ / 128, E_ / 64), 256, 0, stream>>>(
      o16, WpT, bp, x, x1, nullptr, M_, E_, E_, 3, H_, D_, S_);

  // h2 = ln2(x1) -> f16
  ln_kernel<<<M_, 256, 0, stream>>>(x1, ln2_g, ln2_b, nullptr, h2, nullptr, 0);

  // g = gelu(h2 @ W1 + b1) -> f16   (N=3072, K=768)
  gemm_wmma_kernel<<<dim3(M_ / 128, F_ / 64), 256, 0, stream>>>(
      h2, W1T, b1, nullptr, nullptr, g16, M_, F_, E_, 4, H_, D_, S_);

  // f = g @ W2 + b2 -> f32          (N=768, K=3072)
  gemm_wmma_kernel<<<dim3(M_ / 128, E_ / 64), 256, 0, stream>>>(
      g16, W2T, b2, nullptr, fbuf, nullptr, M_, E_, F_, 5, H_, D_, S_);

  // out = x1 + lnf(f)
  ln_kernel<<<M_, 256, 0, stream>>>(fbuf, lnf_g, lnf_b, x1, nullptr, out, 1);
}
